// SelfAttention_62337155334790
// MI455X (gfx1250) — matrix operand order repaired and checked by reading, not compiled, not run on hardware
//
#include <hip/hip_runtime.h>
#include <hip/hip_bf16.h>

// Self-attention forward for MI455X (gfx1250, wave32, WMMA).
// N=2, L=2048, E=1024, H=16, D=64. All matmuls use v_wmma_f32_16x16x32_bf16
// (bf16 operands, f32 accumulate); softmax is an online (flash) softmax in f32.
// K/V tiles are streamed global->LDS with GLOBAL_LOAD_ASYNC_TO_LDS_B128
// (double-buffered, ASYNCcnt-tracked).

#define NB        2
#define LSEQ      2048
#define EMB       1024
#define NHEAD     16
#define HD        64
#define ATT_SCALE 0.03125f   // 1/sqrt(EMB) = 1/32

typedef __attribute__((ext_vector_type(16))) __bf16 v16bf;
typedef __attribute__((ext_vector_type(8)))  float  v8f;

// Pointee type the async-LDS builtin expects: int __attribute__((vector_size(16)))
typedef int v4i_b128 __attribute__((vector_size(16)));
typedef __attribute__((address_space(1))) v4i_b128* gptr_b128;   // global
typedef __attribute__((address_space(3))) v4i_b128* lptr_b128;   // LDS

static __device__ __forceinline__ v8f wmma_bf16(v16bf a, v16bf b, v8f c) {
  // (neg_a, A, neg_b, B, c_mod, C, reuse_a, reuse_b)
  return __builtin_amdgcn_wmma_f32_16x16x32_bf16(false, a, false, b, (short)0, c,
                                                 false, false);
}

// ---- async global->LDS copy (16 bytes per lane) -----------------------------
static __device__ __forceinline__ void async_copy16(const __bf16* gsrc, __bf16* ldst) {
#if __has_builtin(__builtin_amdgcn_global_load_async_to_lds_b128)
  __builtin_amdgcn_global_load_async_to_lds_b128(
      (gptr_b128)(void*)gsrc, (lptr_b128)(void*)ldst,
      /*offset=*/0, /*cpol=*/0);
#else
  *(uint4*)ldst = *(const uint4*)gsrc;
#endif
}

static __device__ __forceinline__ void wait_async0() {
#if __has_builtin(__builtin_amdgcn_s_wait_asynccnt)
  __builtin_amdgcn_s_wait_asynccnt(0);
#else
  asm volatile("s_wait_asynccnt 0x0" ::: "memory");
#endif
}

// ---- WMMA fragment loaders (contiguous k order shared by A and B, wave32) -------------
// A (16x32): lane&15 = M row; element i -> k = i + 16*(lane>>4).
static __device__ __forceinline__ v16bf load_afrag(const __bf16* base, int ld,
                                                   int row, int kOff, int lane) {
  v16bf a;
  const int lh = (lane >> 4) & 1;
#pragma unroll
  for (int i = 0; i < 16; ++i) {
    const int k = kOff + lh * 16 + i;
    a[i] = base[(size_t)row * ld + k];
  }
  return a;
}

// B (32x16, K x N) from a row-major [K][ld] array: lane&15 = N col;
// element i -> k = i + 16*(lane>>4).
static __device__ __forceinline__ v16bf load_bfrag(const __bf16* base, int ld,
                                                   int kOff, int nOff, int lane) {
  v16bf b;
  const int lh = (lane >> 4) & 1;
  const int n  = nOff + (lane & 15);
#pragma unroll
  for (int i = 0; i < 16; ++i)
    b[i] = base[(size_t)(kOff + lh * 16 + i) * ld + n];
  return b;
}

// B (32x16, K x N) from a row-major [N][ld] array (transposed read; the 16
// elements are contiguous in LDS so these vectorize to ds_load_b128).
static __device__ __forceinline__ v16bf load_bfragT(const __bf16* base, int ld,
                                                    int kOff, int nOff, int lane) {
  v16bf b;
  const int lh = (lane >> 4) & 1;
  const int n  = nOff + (lane & 15);
#pragma unroll
  for (int i = 0; i < 16; ++i) {
    const int k = kOff + lh * 16 + i;
    b[i] = base[(size_t)n * ld + k];
  }
  return b;
}

// ---- Kernel 1: per-head projection  Y[r,e] = sum_d X[r,d] * W[e,d] ----------
// Rows r = (n,l,h) flattened (65536 rows x 64), W is [64,64] f32.
// 8 waves/block, each wave computes a 16x64 output tile. Output bf16.
__global__ __launch_bounds__(256) void proj_kernel(const float* __restrict__ X,
                                                   const float* __restrict__ W,
                                                   __bf16* __restrict__ Y) {
  const int lane = threadIdx.x & 31;
  const int wave = threadIdx.x >> 5;
  const int lh   = lane >> 4;
  const int n16  = lane & 15;
  const size_t r0 = ((size_t)blockIdx.x * 8 + wave) * 16;

  // B = W^T (k=d, n=e): b[i] = W[e*64 + d]
  v16bf bfr[4][2];
#pragma unroll
  for (int t = 0; t < 4; ++t)
#pragma unroll
    for (int ks = 0; ks < 2; ++ks) {
      const int e = t * 16 + n16;
#pragma unroll
      for (int i = 0; i < 16; ++i) {
        const int d = ks * 32 + lh * 16 + i;
        bfr[t][ks][i] = (__bf16)W[e * 64 + d];
      }
    }

  // A from X (f32 -> bf16), row = r0 + (lane&15)
  v16bf afr[2];
  const float* xr = X + (r0 + n16) * 64;
#pragma unroll
  for (int ks = 0; ks < 2; ++ks)
#pragma unroll
    for (int i = 0; i < 16; ++i) {
      const int d = ks * 32 + lh * 16 + i;
      afr[ks][i] = (__bf16)xr[d];
    }

  const v8f z8 = {0.f, 0.f, 0.f, 0.f, 0.f, 0.f, 0.f, 0.f};
  v8f acc[4] = {z8, z8, z8, z8};
#pragma unroll
  for (int t = 0; t < 4; ++t)
#pragma unroll
    for (int ks = 0; ks < 2; ++ks)
      acc[t] = wmma_bf16(afr[ks], bfr[t][ks], acc[t]);

#pragma unroll
  for (int t = 0; t < 4; ++t)
#pragma unroll
    for (int r = 0; r < 8; ++r) {
      const size_t row = r0 + r + 8 * lh;
      Y[row * 64 + t * 16 + n16] = (__bf16)acc[t][r];
    }
}

// ---- Kernel 2: flash attention core -----------------------------------------
// Block = 128 threads (4 waves). Block handles 64 query rows of one (n,h);
// wave w owns rows q0+16w..q0+16w+15. Loop over 32 key tiles of 64 with
// double-buffered async K/V staging in LDS.
__global__ __launch_bounds__(128) void attn_kernel(const __bf16* __restrict__ Qp,
                                                   const __bf16* __restrict__ Kp,
                                                   const __bf16* __restrict__ Vp,
                                                   const int* __restrict__ mask,
                                                   __bf16* __restrict__ O) {
  alignas(16) __shared__ __bf16 Ks[2][64 * 64];    // [buf][key][d]
  alignas(16) __shared__ __bf16 Vs[2][64 * 64];    // [buf][key][d]
  alignas(16) __shared__ __bf16 Ps[4][16 * 64];    // per-wave P staging [q][key]

  const int tid  = threadIdx.x;
  const int lane = tid & 31;
  const int wave = tid >> 5;
  const int lh   = lane >> 4;
  const int n16  = lane & 15;
  const int bn   = blockIdx.y / NHEAD;   // batch
  const int h    = blockIdx.y % NHEAD;   // head
  const int q0   = blockIdx.x * 64;
  const size_t base0 = (size_t)bn * LSEQ * EMB + h * HD;   // head base in Kp/Vp

  // Q fragments (16x64 per wave -> two 16x32 A frags), kept in registers.
  v16bf qfr[2];
  {
    const __bf16* qrow =
        Qp + base0 + (size_t)(q0 + wave * 16 + n16) * EMB;
#pragma unroll
    for (int ks = 0; ks < 2; ++ks)
#pragma unroll
      for (int i = 0; i < 16; ++i) {
        const int d = ks * 32 + lh * 16 + i;
        qfr[ks][i] = qrow[d];
      }
  }

  const v8f z8 = {0.f, 0.f, 0.f, 0.f, 0.f, 0.f, 0.f, 0.f};
  v8f acc[4] = {z8, z8, z8, z8};
  float mOld[8], lSum[8];
#pragma unroll
  for (int r = 0; r < 8; ++r) { mOld[r] = -3.4e38f; lSum[r] = 0.f; }

  // Cooperative K/V tile issue: thread t handles row t/2, 32-elem half t%2;
  // 4x 16-byte async copies per matrix per thread (8 KB per tile).
  const int crow = tid >> 1;
  const int chh  = (tid & 1) * 32;

  // Preload tile 0 into buffer 0.
  {
    const size_t src = base0 + (size_t)crow * EMB + chh;
#pragma unroll
    for (int i = 0; i < 4; ++i) {
      async_copy16(Kp + src + i * 8, &Ks[0][crow * 64 + chh + i * 8]);
      async_copy16(Vp + src + i * 8, &Vs[0][crow * 64 + chh + i * 8]);
    }
  }
  wait_async0();
  __syncthreads();

  const int NT = LSEQ / 64;
  for (int kb = 0; kb < NT; ++kb) {
    const int k0  = kb * 64;
    const int cur = kb & 1;

    // Issue async loads for the next tile into the idle buffer.
    if (kb + 1 < NT) {
      const size_t src = base0 + (size_t)(k0 + 64 + crow) * EMB + chh;
#pragma unroll
      for (int i = 0; i < 4; ++i) {
        async_copy16(Kp + src + i * 8, &Ks[cur ^ 1][crow * 64 + chh + i * 8]);
        async_copy16(Vp + src + i * 8, &Vs[cur ^ 1][crow * 64 + chh + i * 8]);
      }
      // Prefetch next tile's mask block (64 rows x 256 B).
      __builtin_prefetch(
          &mask[((size_t)bn * LSEQ + q0 + crow) * LSEQ + k0 + 64 + chh], 0, 0);
    }

    // S(16x64) = Q(16x64) * K^T : B[k=d][n=key] = Ks[key][d] (transposed read)
    v8f S[4] = {z8, z8, z8, z8};
#pragma unroll
    for (int t = 0; t < 4; ++t)
#pragma unroll
      for (int ks = 0; ks < 2; ++ks) {
        v16bf bk = load_bfragT(Ks[cur], 64, ks * 32, t * 16, lane);
        S[t] = wmma_bf16(qfr[ks], bk, S[t]);
      }

    // Mask (pre-scale -1e20, like reference), scale, per-row running max.
    float rmax[8];
#pragma unroll
    for (int r = 0; r < 8; ++r) rmax[r] = -3.4e38f;
#pragma unroll
    for (int t = 0; t < 4; ++t) {
      const int col = k0 + t * 16 + n16;
#pragma unroll
      for (int r = 0; r < 8; ++r) {
        const int qr = q0 + wave * 16 + r + 8 * lh;
        const int mv = mask[((size_t)bn * LSEQ + qr) * LSEQ + col];
        const float s = (mv != 0) ? (S[t][r] * ATT_SCALE) : -3.125e18f;
        S[t][r] = s;
        rmax[r] = fmaxf(rmax[r], s);
      }
    }
    // A row lives across the 16 lanes of one half -> xor-reduce within halves.
#pragma unroll
    for (int r = 0; r < 8; ++r)
#pragma unroll
      for (int off = 1; off < 16; off <<= 1)
        rmax[r] = fmaxf(rmax[r], __shfl_xor(rmax[r], off, 32));

    float corr[8], psum[8];
#pragma unroll
    for (int r = 0; r < 8; ++r) {
      const float mN = fmaxf(mOld[r], rmax[r]);
      corr[r] = __expf(mOld[r] - mN);
      mOld[r] = mN;
      psum[r] = 0.f;
    }
#pragma unroll
    for (int t = 0; t < 4; ++t)
#pragma unroll
      for (int r = 0; r < 8; ++r) {
        const float p = __expf(S[t][r] - mOld[r]);
        S[t][r] = p;
        psum[r] += p;
      }
#pragma unroll
    for (int r = 0; r < 8; ++r) {
#pragma unroll
      for (int off = 1; off < 16; off <<= 1)
        psum[r] += __shfl_xor(psum[r], off, 32);
      lSum[r] = lSum[r] * corr[r] + psum[r];
    }
#pragma unroll
    for (int t = 0; t < 4; ++t)
#pragma unroll
      for (int r = 0; r < 8; ++r)
        acc[t][r] *= corr[r];

    // Transpose P (D layout -> A layout) through per-wave LDS staging.
    __bf16* pw = Ps[wave];
#pragma unroll
    for (int t = 0; t < 4; ++t)
#pragma unroll
      for (int r = 0; r < 8; ++r)
        pw[(r + 8 * lh) * 64 + t * 16 + n16] = (__bf16)S[t][r];
    __syncthreads();

    v16bf pa[2];
#pragma unroll
    for (int ks = 0; ks < 2; ++ks)
      pa[ks] = load_afrag(pw, 64, n16, ks * 32, lane);

    // acc(16x64) += P(16x64) * V(64x64): B[k=key][n=d] = Vs[key][d]
#pragma unroll
    for (int t = 0; t < 4; ++t)
#pragma unroll
      for (int ks = 0; ks < 2; ++ks) {
        v16bf bv = load_bfrag(Vs[cur], 64, ks * 32, t * 16, lane);
        acc[t] = wmma_bf16(pa[ks], bv, acc[t]);
      }

    // Next tile's async copies must have landed before anyone computes on it,
    // and all waves must be done with this tile before it is overwritten.
    wait_async0();
    __syncthreads();
  }

  float invl[8];
#pragma unroll
  for (int r = 0; r < 8; ++r) invl[r] = 1.0f / lSum[r];
#pragma unroll
  for (int t = 0; t < 4; ++t)
#pragma unroll
    for (int r = 0; r < 8; ++r) {
      const int row = q0 + wave * 16 + r + 8 * lh;
      O[(size_t)(bn * LSEQ + row) * EMB + h * HD + t * 16 + n16] =
          (__bf16)(acc[t][r] * invl[r]);
    }
}

// ---- Kernel 3: output projection  out = O @ Wout^T + bout -------------------
// M=4096, N=1024, K=1024. Block = 128 threads (4 waves) -> 64x64 output tile.
__global__ __launch_bounds__(128) void outproj_kernel(const __bf16* __restrict__ O,
                                                      const float* __restrict__ Wout,
                                                      const float* __restrict__ bout,
                                                      float* __restrict__ out) {
  alignas(16) __shared__ __bf16 Ws[32 * 64];   // [k][e] tile of Wout^T
  const int tid  = threadIdx.x;
  const int lane = tid & 31;
  const int wave = tid >> 5;
  const int lh   = lane >> 4;
  const int n16  = lane & 15;
  const size_t m0 = (size_t)blockIdx.x * 64 + wave * 16;
  const int e0 = blockIdx.y * 64;

  const v8f z8 = {0.f, 0.f, 0.f, 0.f, 0.f, 0.f, 0.f, 0.f};
  v8f acc[4] = {z8, z8, z8, z8};
  const __bf16* orow = O + (m0 + n16) * EMB;

  for (int kb = 0; kb < EMB / 32; ++kb) {
    const int kOff = kb * 32;
    // Stage Wout^T tile: Ws[k][e] = Wout[(e0+e)*EMB + kOff+k], f32 -> bf16.
    for (int i = tid; i < 32 * 64; i += 128) {
      const int kk = i >> 6;
      const int ee = i & 63;
      Ws[kk * 64 + ee] = (__bf16)Wout[(size_t)(e0 + ee) * EMB + kOff + kk];
    }
    __syncthreads();

    v16bf a;
#pragma unroll
    for (int i = 0; i < 16; ++i) {
      const int k = kOff + lh * 16 + i;
      a[i] = orow[k];
    }
#pragma unroll
    for (int t = 0; t < 4; ++t) {
      v16bf bb = load_bfrag(Ws, 64, 0, t * 16, lane);
      acc[t] = wmma_bf16(a, bb, acc[t]);
    }
    __syncthreads();
  }

#pragma unroll
  for (int t = 0; t < 4; ++t)
#pragma unroll
    for (int r = 0; r < 8; ++r) {
      const size_t row = m0 + r + 8 * lh;
      const int e = e0 + t * 16 + n16;
      out[row * EMB + e] = acc[t][r] + bout[e];
    }
}

// ---- Host launch ------------------------------------------------------------
extern "C" void kernel_launch(void* const* d_in, const int* in_sizes, int n_in,
                              void* d_out, int out_size, void* d_ws, size_t ws_size,
                              hipStream_t stream) {
  (void)in_sizes; (void)n_in; (void)out_size; (void)ws_size;
  const float* values = (const float*)d_in[0];
  const float* keysI  = (const float*)d_in[1];
  const float* query  = (const float*)d_in[2];
  const int*   mask   = (const int*)d_in[3];
  const float* Wv     = (const float*)d_in[4];
  const float* Wk     = (const float*)d_in[5];
  const float* Wq     = (const float*)d_in[6];
  const float* Wout   = (const float*)d_in[7];
  const float* bout   = (const float*)d_in[8];
  float* out = (float*)d_out;

  const size_t elems = (size_t)NB * LSEQ * EMB;   // 4 Mi elements per buffer
  __bf16* Qp = (__bf16*)d_ws;
  __bf16* Kp = Qp + elems;
  __bf16* Vp = Kp + elems;
  __bf16* Op = Vp + elems;

  // 1) Per-head projections (65536 rows of 64): 512 blocks x 8 waves x 16 rows.
  proj_kernel<<<dim3(512), 256, 0, stream>>>(query,  Wq, Qp);
  proj_kernel<<<dim3(512), 256, 0, stream>>>(keysI,  Wk, Kp);
  proj_kernel<<<dim3(512), 256, 0, stream>>>(values, Wv, Vp);

  // 2) Flash attention: grid (L/64 query tiles, N*H).
  attn_kernel<<<dim3(LSEQ / 64, NB * NHEAD), 128, 0, stream>>>(Qp, Kp, Vp, mask, Op);

  // 3) Output projection: grid (4096/64 row tiles, 1024/64 col tiles).
  outproj_kernel<<<dim3((NB * LSEQ) / 64, EMB / 64), 128, 0, stream>>>(Op, Wout,
                                                                      bout, out);
}
